// HFLlavaViTAttention_80513456931378
// MI455X (gfx1250) — matrix operand order repaired and checked by reading, not compiled, not run on hardware
//
#include <hip/hip_runtime.h>
#include <hip/hip_bf16.h>

// ---------------------------------------------------------------------------
// Types for WMMA fragments (gfx1250 wave32)
// ---------------------------------------------------------------------------
typedef __attribute__((ext_vector_type(16))) __bf16 bf16x16;
typedef __attribute__((ext_vector_type(8)))  float  f32x8;

union AFrag {
    bf16x16        v;
    unsigned short u[16];
    uint4          q[2];
};

__device__ __forceinline__ unsigned short f2bf(float f) {
    unsigned u = __float_as_uint(f);
    u += 0x7FFFu + ((u >> 16) & 1u);   // round-to-nearest-even
    return (unsigned short)(u >> 16);
}

#define WMMA_BF16(A, B, C) \
    __builtin_amdgcn_wmma_f32_16x16x32_bf16(false, (A), false, (B), (short)0, (C), false, false)

// Problem constants
#define BB 16
#define SS 1024
#define DD 1024
#define HH 16
#define HD 64

// ---------------------------------------------------------------------------
// Kernel 0: fp32 -> bf16 conversion (weights + activations), 4 elems/thread
// ---------------------------------------------------------------------------
__global__ __launch_bounds__(256)
void cvt_f32_to_bf16_x4(const float* __restrict__ in,
                        unsigned short* __restrict__ out, int n4) {
    int i = blockIdx.x * 256 + threadIdx.x;
    if (i >= n4) return;
    float4 f = ((const float4*)in)[i];
    uint2 u;
    u.x = (unsigned)f2bf(f.x) | ((unsigned)f2bf(f.y) << 16);
    u.y = (unsigned)f2bf(f.z) | ((unsigned)f2bf(f.w) << 16);
    ((uint2*)out)[i] = u;
}

// ---------------------------------------------------------------------------
// Kernel 1: fused QKV projection + bias + RoPE, one wave per (16-row, head)
//   A (bf16 hidden) loaded once per k-step, reused for 12 WMMAs (q,k,v x 4)
//   Q -> [B,H,S,HD] bf16, K -> [B,H,S,HD] bf16, V -> [B,H,HD,S] bf16 (transposed)
// ---------------------------------------------------------------------------
__global__ __launch_bounds__(32)
void qkv_rope_kernel(const unsigned short* __restrict__ x,  // [B*S, D] bf16
                     const float* __restrict__ rope,        // [B*S, HD] fp32
                     const unsigned short* __restrict__ qw, // [D,D] bf16
                     const unsigned short* __restrict__ kw,
                     const unsigned short* __restrict__ vw,
                     const float* __restrict__ qb,
                     const float* __restrict__ kb,
                     const float* __restrict__ vb,
                     unsigned short* __restrict__ qo,
                     unsigned short* __restrict__ ko,
                     unsigned short* __restrict__ vto) {
    const int lane  = threadIdx.x;
    const int half  = lane >> 4;     // 0 or 1
    const int l16   = lane & 15;
    const int mbase = blockIdx.x * 16;
    const int h     = blockIdx.y;
    const int nb    = h * HD;

    const unsigned short* wbf[3]  = {qw, kw, vw};
    const float*          bias[3] = {qb, kb, vb};

    f32x8 acc[3][4] = {};

    const unsigned short* aprow = x + (size_t)(mbase + l16) * DD + half * 8;

    for (int k = 0; k < DD; k += 32) {
        // A fragment: chunks of 8 at k0 and k0+16 (ISA 16-bit A layout)
        AFrag a;
        a.q[0] = *(const uint4*)(aprow + k);
        a.q[1] = *(const uint4*)(aprow + k + 16);

#pragma unroll
        for (int p = 0; p < 3; p++) {
#pragma unroll
            for (int t = 0; t < 4; t++) {
                // B fragment: column n = weight row (contiguous in k)
                const unsigned short* bp =
                    wbf[p] + (size_t)(nb + t * 16 + l16) * DD + k + half * 8;
                AFrag bfr;
                bfr.q[0] = *(const uint4*)(bp);
                bfr.q[1] = *(const uint4*)(bp + 16);
                acc[p][t] = WMMA_BF16(a.v, bfr.v, acc[p][t]);
            }
        }
    }

    // bias (column quantity in C/D layout)
#pragma unroll
    for (int p = 0; p < 3; p++)
#pragma unroll
        for (int t = 0; t < 4; t++) {
            float bv = bias[p][nb + t * 16 + l16];
#pragma unroll
            for (int r = 0; r < 8; r++) acc[p][t][r] += bv;
        }

    // RoPE on Q and K: hd pairs (t, t±2) are same-lane/same-slot.
    // Angles depend only on (row, hd) -> computed once, applied to both Q,K.
#pragma unroll
    for (int r = 0; r < 8; r++) {
        int m = mbase + r + half * 8;              // flat (b,s)
        const float* rp = rope + (size_t)m * HD;
        float f0 = rp[0 * 16 + l16], f1 = rp[1 * 16 + l16];
        float f2 = rp[2 * 16 + l16], f3 = rp[3 * 16 + l16];
        float c0 = __cosf(f0), s0 = __sinf(f0);
        float c1 = __cosf(f1), s1 = __sinf(f1);
        float c2 = __cosf(f2), s2 = __sinf(f2);
        float c3 = __cosf(f3), s3 = __sinf(f3);
#pragma unroll
        for (int p = 0; p < 2; p++) {
            float o0 = acc[p][0][r], o1 = acc[p][1][r];
            float o2 = acc[p][2][r], o3 = acc[p][3][r];
            acc[p][0][r] = o0 * c0 - o2 * s0;
            acc[p][1][r] = o1 * c1 - o3 * s1;
            acc[p][2][r] = o2 * c2 + o0 * s2;
            acc[p][3][r] = o3 * c3 + o1 * s3;
        }
    }

    // stores
#pragma unroll
    for (int t = 0; t < 4; t++) {
        int hd = t * 16 + l16;
#pragma unroll
        for (int r = 0; r < 8; r++) {
            int m  = mbase + r + half * 8;
            int b  = m >> 10, s = m & (SS - 1);
            int bh = b * HH + h;
            qo[((size_t)bh * SS + s) * HD + hd]  = f2bf(acc[0][t][r]);
            ko[((size_t)bh * SS + s) * HD + hd]  = f2bf(acc[1][t][r]);
            vto[((size_t)bh * HD + hd) * SS + s] = f2bf(acc[2][t][r]);
        }
    }
}

// ---------------------------------------------------------------------------
// Kernel 2: flash attention, one wave per (b, h, 16-row q-tile); k-step = 32
// ---------------------------------------------------------------------------
__global__ __launch_bounds__(32)
void flash_attn_kernel(const unsigned short* __restrict__ q,
                       const unsigned short* __restrict__ k,
                       const unsigned short* __restrict__ vt,
                       unsigned short* __restrict__ ctx) {   // [B,S,D] bf16
    __shared__ unsigned short pls[16 * 32];   // P tile (row-major 16x32)

    const int lane = threadIdx.x;
    const int half = lane >> 4, l16 = lane & 15;
    const int qt = blockIdx.x, h = blockIdx.y, b = blockIdx.z;
    const int bh = b * HH + h;
    const size_t base = (size_t)bh * SS * HD;

    // Q A-fragments (hd segments 0..31 and 32..63), loaded once
    AFrag qa[2];
    {
        const unsigned short* qp = q + base + (size_t)(qt * 16 + l16) * HD;
        int k0 = half * 8;
#pragma unroll
        for (int c = 0; c < 2; c++) {
            qa[c].q[0] = *(const uint4*)(qp + c * 32 + k0);
            qa[c].q[1] = *(const uint4*)(qp + c * 32 + k0 + 16);
        }
    }

    float m_i[8], l_i[8];
    f32x8 o[4] = {};
#pragma unroll
    for (int r = 0; r < 8; r++) { m_i[r] = -1e30f; l_i[r] = 0.f; }

    const float scale = 0.125f;   // HD^-0.5

    for (int j = 0; j < SS; j += 32) {
        // scores for two 16-col tiles
        f32x8 s0 = {}, s1 = {};
#pragma unroll
        for (int jj = 0; jj < 2; jj++) {
            f32x8 sacc = {};
#pragma unroll
            for (int c = 0; c < 2; c++) {
                const unsigned short* kp =
                    k + base + (size_t)(j + jj * 16 + l16) * HD + c * 32 + half * 8;
                AFrag kb;
                kb.q[0] = *(const uint4*)(kp);
                kb.q[1] = *(const uint4*)(kp + 16);
                sacc = WMMA_BF16(qa[c].v, kb.v, sacc);
            }
            if (jj == 0) s0 = sacc; else s1 = sacc;
        }

        // online softmax (row = C-layout vgpr slot; reduce across 16 lanes)
        float alpha[8];
#pragma unroll
        for (int r = 0; r < 8; r++) {
            float sa = s0[r] * scale, sb = s1[r] * scale;
            float rm = fmaxf(sa, sb);
            rm = fmaxf(rm, __shfl_xor(rm, 1, 32));
            rm = fmaxf(rm, __shfl_xor(rm, 2, 32));
            rm = fmaxf(rm, __shfl_xor(rm, 4, 32));
            rm = fmaxf(rm, __shfl_xor(rm, 8, 32));
            float mn = fmaxf(m_i[r], rm);
            float al = __expf(m_i[r] - mn);
            float p0 = __expf(sa - mn), p1 = __expf(sb - mn);
            float rs = p0 + p1;
            rs += __shfl_xor(rs, 1, 32);
            rs += __shfl_xor(rs, 2, 32);
            rs += __shfl_xor(rs, 4, 32);
            rs += __shfl_xor(rs, 8, 32);
            l_i[r]  = l_i[r] * al + rs;
            m_i[r]  = mn;
            alpha[r] = al;
            s0[r] = p0; s1[r] = p1;
        }

        // C-layout -> A-fragment reshape through LDS
        __syncthreads();
#pragma unroll
        for (int r = 0; r < 8; r++) {
            int m = r + half * 8;
            pls[m * 32 + l16]      = f2bf(s0[r]);
            pls[m * 32 + 16 + l16] = f2bf(s1[r]);
        }
        __syncthreads();
        AFrag pa;
        {
            const unsigned short* pp = pls + l16 * 32 + half * 8;
            pa.q[0] = *(const uint4*)(pp);
            pa.q[1] = *(const uint4*)(pp + 16);
        }

        // O update: rescale + P x V (Vt rows contiguous in seq)
#pragma unroll
        for (int t = 0; t < 4; t++) {
#pragma unroll
            for (int r = 0; r < 8; r++) o[t][r] *= alpha[r];
            const unsigned short* vp =
                vt + base + (size_t)(t * 16 + l16) * SS + j + half * 8;
            AFrag vb;
            vb.q[0] = *(const uint4*)(vp);
            vb.q[1] = *(const uint4*)(vp + 16);
            o[t] = WMMA_BF16(pa.v, vb.v, o[t]);
        }
    }

    // finalize and store context in [B,S,D] layout (D = h*64 + hd)
#pragma unroll
    for (int r = 0; r < 8; r++) {
        float inv = 1.f / l_i[r];
        int s = qt * 16 + r + half * 8;
        size_t rowoff = ((size_t)b * SS + s) * DD + h * HD;
#pragma unroll
        for (int t = 0; t < 4; t++)
            ctx[rowoff + t * 16 + l16] = f2bf(o[t][r] * inv);
    }
}

// ---------------------------------------------------------------------------
// Kernel 3: output projection  out = ctx @ o_w.T + o_b  (fp32 out)
// ---------------------------------------------------------------------------
__global__ __launch_bounds__(32)
void out_proj_kernel(const unsigned short* __restrict__ ctx,  // [M, D] bf16
                     const unsigned short* __restrict__ ow,   // [D, D] bf16
                     const float* __restrict__ ob,
                     float* __restrict__ out) {               // [M, D] fp32
    const int lane = threadIdx.x;
    const int half = lane >> 4, l16 = lane & 15;
    const int mbase = blockIdx.x * 16;
    const int nbase = blockIdx.y * 64;

    f32x8 acc[4] = {};
    const unsigned short* ap0 = ctx + (size_t)(mbase + l16) * DD + half * 8;

    for (int k = 0; k < DD; k += 32) {
        AFrag a;
        a.q[0] = *(const uint4*)(ap0 + k);
        a.q[1] = *(const uint4*)(ap0 + k + 16);
#pragma unroll
        for (int t = 0; t < 4; t++) {
            const unsigned short* bp =
                ow + (size_t)(nbase + t * 16 + l16) * DD + k + half * 8;
            AFrag bfr;
            bfr.q[0] = *(const uint4*)(bp);
            bfr.q[1] = *(const uint4*)(bp + 16);
            acc[t] = WMMA_BF16(a.v, bfr.v, acc[t]);
        }
    }
#pragma unroll
    for (int t = 0; t < 4; t++) {
        int col = nbase + t * 16 + l16;
        float bv = ob[col];
#pragma unroll
        for (int r = 0; r < 8; r++) {
            int row = mbase + r + half * 8;
            out[(size_t)row * DD + col] = acc[t][r] + bv;
        }
    }
}

// ---------------------------------------------------------------------------
// Host launch
// ---------------------------------------------------------------------------
extern "C" void kernel_launch(void* const* d_in, const int* in_sizes, int n_in,
                              void* d_out, int out_size, void* d_ws, size_t ws_size,
                              hipStream_t stream) {
    const float* x    = (const float*)d_in[0];   // [B,S,D]
    const float* rope = (const float*)d_in[1];   // [B,S,HD]
    const float* q_w  = (const float*)d_in[2];
    const float* q_b  = (const float*)d_in[3];
    const float* k_w  = (const float*)d_in[4];
    const float* k_b  = (const float*)d_in[5];
    const float* v_w  = (const float*)d_in[6];
    const float* v_b  = (const float*)d_in[7];
    const float* o_w  = (const float*)d_in[8];
    const float* o_b  = (const float*)d_in[9];
    float* out = (float*)d_out;

    // workspace carving (bf16 stored as ushort)
    const size_t T = (size_t)BB * HH * SS * HD;   // 16,777,216
    const size_t W = (size_t)DD * DD;             //  1,048,576
    unsigned short* ws    = (unsigned short*)d_ws;
    unsigned short* qws   = ws;
    unsigned short* kws   = ws + T;
    unsigned short* vtws  = ws + 2 * T;
    unsigned short* ctxws = ws + 3 * T;
    unsigned short* xbf   = ws + 4 * T;
    unsigned short* qwbf  = ws + 5 * T;
    unsigned short* kwbf  = qwbf + W;
    unsigned short* vwbf  = kwbf + W;
    unsigned short* owbf  = vwbf + W;

    // 1) convert activations + weights to bf16
    {
        int nx4 = (int)(T / 4);                       // hidden_states
        cvt_f32_to_bf16_x4<<<(nx4 + 255) / 256, 256, 0, stream>>>(x, xbf, nx4);
        int n4 = (int)(W / 4);
        int blocks = (n4 + 255) / 256;
        cvt_f32_to_bf16_x4<<<blocks, 256, 0, stream>>>(q_w, qwbf, n4);
        cvt_f32_to_bf16_x4<<<blocks, 256, 0, stream>>>(k_w, kwbf, n4);
        cvt_f32_to_bf16_x4<<<blocks, 256, 0, stream>>>(v_w, vwbf, n4);
        cvt_f32_to_bf16_x4<<<blocks, 256, 0, stream>>>(o_w, owbf, n4);
    }
    // 2) fused QKV + RoPE
    qkv_rope_kernel<<<dim3(BB * SS / 16, HH), 32, 0, stream>>>(
        xbf, rope, qwbf, kwbf, vwbf, q_b, k_b, v_b, qws, kws, vtws);
    // 3) flash attention
    flash_attn_kernel<<<dim3(SS / 16, HH, BB), 32, 0, stream>>>(
        qws, kws, vtws, ctxws);
    // 4) output projection
    out_proj_kernel<<<dim3(BB * SS / 16, DD / 64), 32, 0, stream>>>(
        ctxws, owbf, o_b, out);

    (void)in_sizes; (void)n_in; (void)out_size; (void)ws_size;
}
